// DepthAwareTokenAggregator_33913061769566
// MI455X (gfx1250) — matrix operand order repaired and checked by reading, not compiled, not run on hardware
//
#include <hip/hip_runtime.h>
#include <hip/hip_bf16.h>
#include <math.h>

#define BDIM 8
#define NDIM 512
#define YDIM 64
#define CDIM 256

#define TSTRIDE 260   // fp32 tile row stride (floats), padded
#define ASTRIDE 264   // bf16 tile row stride (bf16 elems), padded, *2 bytes = 16B multiple
#define WSTRIDE 264   // W1T row stride (bf16 elems), padded, *2 bytes = 16B multiple

typedef __attribute__((ext_vector_type(16))) __bf16 bf16x16;
typedef __attribute__((ext_vector_type(8)))  float  f32x8;
typedef int v4i __attribute__((vector_size(16)));   // matches builtin's expected param type

// ---- async-to-LDS availability probes ----
#ifndef __has_builtin
#define __has_builtin(x) 0
#endif
#if __has_builtin(__builtin_amdgcn_global_load_async_to_lds_b128)
#define HAVE_ASYNC_LDS 1
#else
#define HAVE_ASYNC_LDS 0
#endif
#if __has_builtin(__builtin_amdgcn_s_wait_asynccnt)
#define WAIT_ASYNC0() __builtin_amdgcn_s_wait_asynccnt(0)
#else
#define WAIT_ASYNC0() asm volatile("s_wait_asynccnt 0x0" ::: "memory")
#endif

// ---- LDS carve (bytes) ----
#define OFF_TILE   0
#define SZ_TILE    (YDIM * TSTRIDE * 4)          // 66560
#define OFF_TILEB  (OFF_TILE + SZ_TILE)          // 66560
#define SZ_TILEB   (YDIM * ASTRIDE * 2)          // 33792
#define OFF_W1T    (OFF_TILEB + SZ_TILEB)        // 100352 (16B aligned)
#define SZ_W1T     (CDIM * WSTRIDE * 2)          // 135168
#define OFF_B1     (OFF_W1T + SZ_W1T)            // 235520
#define OFF_W257   (OFF_B1   + CDIM * 4)
#define OFF_W2     (OFF_W257 + CDIM * 4)
#define OFF_LOGIT  (OFF_W2   + CDIM * 4)
#define OFF_VALID  (OFF_LOGIT + YDIM * 4)
#define OFF_YNORM  (OFF_VALID + YDIM * 4)
#define OFF_EX     (OFF_YNORM + YDIM * 4)
#define OFF_WV     (OFF_EX    + YDIM * 4)
#define OFF_WT     (OFF_WV    + YDIM * 4)
#define SMEM_BYTES (OFF_WT    + YDIM * 4)        // 240128 bytes < 320KB/WGP

__global__ __launch_bounds__(256, 1)
void datok_kernel(const float* __restrict__ sampled,   // [B,N,Y,C]
                  const float* __restrict__ valid,     // [B,N,Y]
                  const float* __restrict__ ynorm,     // [Y]
                  const float* __restrict__ W1,        // [C+1, C]
                  const float* __restrict__ b1,        // [C]
                  const float* __restrict__ W2,        // [C,1]
                  const float* __restrict__ b2,        // [1]
                  float* __restrict__ out)             // [B,N,C]
{
    extern __shared__ char smem[];
    float*  tile    = (float*) (smem + OFF_TILE);
    __bf16* tileb   = (__bf16*)(smem + OFF_TILEB);
    __bf16* w1t     = (__bf16*)(smem + OFF_W1T);
    float*  b1f     = (float*) (smem + OFF_B1);
    float*  w257    = (float*) (smem + OFF_W257);
    float*  w2f     = (float*) (smem + OFF_W2);
    float*  logitsL = (float*) (smem + OFF_LOGIT);
    float*  validL  = (float*) (smem + OFF_VALID);
    float*  ynormL  = (float*) (smem + OFF_YNORM);
    float*  exL     = (float*) (smem + OFF_EX);
    float*  wvL     = (float*) (smem + OFF_WV);
    float*  wtL     = (float*) (smem + OFF_WT);

    const int tid = threadIdx.x;
    const int bn  = blockIdx.x;                       // token id: b*N + n
    const float* tsrc = sampled + (size_t)bn * YDIM * CDIM;

    // ---- Phase 0: fp32 sampled tile [64][256] -> padded LDS via ASYNC copy (no VGPR hop) ----
    {
        const float4* src4 = (const float4*)tsrc;
        #pragma unroll
        for (int i = 0; i < 16; ++i) {
            int idx4 = i * 256 + tid;                 // 0..4095 float4s (contiguous global)
            int r = idx4 >> 6;                        // row 0..63
            int c = (idx4 & 63) << 2;                 // col 0..252 step 4
#if HAVE_ASYNC_LDS
            __builtin_amdgcn_global_load_async_to_lds_b128(
                (v4i*)(src4 + idx4),
                (v4i*)(tile + r * TSTRIDE + c), 0, 0);
#else
            float4 v = src4[idx4];
            *(float4*)(tile + r * TSTRIDE + c) = v;
#endif
        }
    }

    // ---- Phase 1 (overlaps async copy): stage W1[0:256,:] transposed bf16, pair-packed K ----
    {
        const float4* w4 = (const float4*)W1;
        #pragma unroll 2
        for (int i = 0; i < 32; ++i) {
            int p  = i * 256 + tid;                   // pair index 0..8191
            int kp = p >> 6;                          // k-pair 0..127 -> k = 2*kp, 2*kp+1
            int n4 = (p & 63) << 2;                   // 4 consecutive output cols
            float4 g0 = w4[(kp * 2 + 0) * 64 + (p & 63)];   // row k,   cols n4..n4+3
            float4 g1 = w4[(kp * 2 + 1) * 64 + (p & 63)];   // row k+1, cols n4..n4+3
            union { __bf16 b[2]; unsigned int u; } pk;
            pk.b[0] = (__bf16)g0.x; pk.b[1] = (__bf16)g1.x;
            *(unsigned int*)(w1t + (n4 + 0) * WSTRIDE + kp * 2) = pk.u;
            pk.b[0] = (__bf16)g0.y; pk.b[1] = (__bf16)g1.y;
            *(unsigned int*)(w1t + (n4 + 1) * WSTRIDE + kp * 2) = pk.u;
            pk.b[0] = (__bf16)g0.z; pk.b[1] = (__bf16)g1.z;
            *(unsigned int*)(w1t + (n4 + 2) * WSTRIDE + kp * 2) = pk.u;
            pk.b[0] = (__bf16)g0.w; pk.b[1] = (__bf16)g1.w;
            *(unsigned int*)(w1t + (n4 + 3) * WSTRIDE + kp * 2) = pk.u;
        }
    }
    // ---- small vectors: b1, W1 row 256 (y-feature weights), W2, valid, y_norm, logits=b2 ----
    if (tid < CDIM) {
        b1f[tid]  = b1[tid];
        w257[tid] = W1[256 * CDIM + tid];
        w2f[tid]  = W2[tid];
    }
    if (tid < YDIM) {
        validL[tid]  = valid[(size_t)bn * YDIM + tid];
        ynormL[tid]  = ynorm[tid];
        logitsL[tid] = b2[0];                         // init logits with output bias
    }

    // ---- Phase 0b: drain async copy, then build bf16 tile (reads only this thread's data) ----
    WAIT_ASYNC0();
    {
        #pragma unroll 4
        for (int i = 0; i < 16; ++i) {
            int idx4 = i * 256 + tid;
            int r = idx4 >> 6;
            int c = (idx4 & 63) << 2;
            float4 v = *(const float4*)(tile + r * TSTRIDE + c);
            union { __bf16 b[4]; unsigned long long u; } pk;
            pk.b[0] = (__bf16)v.x; pk.b[1] = (__bf16)v.y;
            pk.b[2] = (__bf16)v.z; pk.b[3] = (__bf16)v.w;
            *(unsigned long long*)(tileb + r * ASTRIDE + c) = pk.u;
        }
    }
    __syncthreads();

    // ---- Phase 2: h = GELU(tileb @ W1 + bias(y)); logits += h @ W2 (h never stored) ----
    const int wave = tid >> 5, lane = tid & 31;
    const int h  = lane >> 4;                         // K-half selector
    const int cx = lane & 15;                         // row (A) / column (B,C) index
    const int n0 = wave * 32 + cx;                    // wave owns hidden cols [32w, 32w+32)
    const int n1 = n0 + 16;
    const float b1_0 = b1f[n0],  b1_1 = b1f[n1];
    const float w57_0 = w257[n0], w57_1 = w257[n1];
    const float w2_0 = w2f[n0],  w2_1 = w2f[n1];

    #pragma unroll 1
    for (int m = 0; m < 4; ++m) {
        f32x8 acc0 = {0.f,0.f,0.f,0.f,0.f,0.f,0.f,0.f};
        f32x8 acc1 = {0.f,0.f,0.f,0.f,0.f,0.f,0.f,0.f};
        const __bf16* arow = tileb + (16 * m + cx) * ASTRIDE + 8 * h;
        #pragma unroll
        for (int kk = 0; kk < 8; ++kk) {
            // A fragment: 16x32 bf16 — lane = row 16m+cx; K = 32kk + {8h..8h+7, 16+8h..16+8h+7}
            const __bf16* pa = arow + kk * 32;
            union AU { uint4 q[2]; bf16x16 v; } au;
            au.q[0] = *(const uint4*)(pa);            // 8 bf16, K low chunk
            au.q[1] = *(const uint4*)(pa + 16);       // 8 bf16, K high chunk
            // B fragments: 32x16 bf16 — lane = column n; K = 32kk + {8h..8h+7, 16+8h..16+8h+7} (transposed W1)
            const __bf16* pb0 = w1t + n0 * WSTRIDE + kk * 32 + 8 * h;
            const __bf16* pb1 = w1t + n1 * WSTRIDE + kk * 32 + 8 * h;
            union BU { uint4 q[2]; bf16x16 v; } bu0, bu1;
            bu0.q[0] = *(const uint4*)(pb0); bu0.q[1] = *(const uint4*)(pb0 + 16);
            bu1.q[0] = *(const uint4*)(pb1); bu1.q[1] = *(const uint4*)(pb1 + 16);
            acc0 = __builtin_amdgcn_wmma_f32_16x16x32_bf16(false, au.v, false, bu0.v,
                                                           (short)0, acc0, false, false);
            acc1 = __builtin_amdgcn_wmma_f32_16x16x32_bf16(false, au.v, false, bu1.v,
                                                           (short)0, acc1, false, false);
        }
        // epilogue: bias + exact GELU + dot with W2, 16-lane shuffle reduce, ds_add logits
        #pragma unroll
        for (int j = 0; j < 8; ++j) {
            int y = 16 * m + j + 8 * h;               // C/D layout: VGPR j -> row 16m+j+8h
            float yn = ynormL[y];
            float x0 = acc0[j] + b1_0 + yn * w57_0;
            float x1 = acc1[j] + b1_1 + yn * w57_1;
            float g0 = 0.5f * x0 * (1.0f + erff(x0 * 0.70710678118654752f));
            float g1 = 0.5f * x1 * (1.0f + erff(x1 * 0.70710678118654752f));
            float val = g0 * w2_0 + g1 * w2_1;
            #pragma unroll
            for (int off = 1; off < 16; off <<= 1)    // stays within 16-lane half
                val += __shfl_xor(val, off, 32);
            if (cx == 0) atomicAdd(&logitsL[y], val); // lanes 0 and 16 -> ds_add_f32
        }
    }
    __syncthreads();

    // ---- Phase 3: masked softmax over Y, * valid, renormalize ----
    if (tid < YDIM) {
        float v = validL[tid];
        float l = logitsL[tid];
        logitsL[tid] = (v < 0.5f) ? -10000.0f : l;
    }
    __syncthreads();
    if (tid < YDIM) {
        float mx = -3.4e38f;
        for (int y = 0; y < YDIM; ++y) mx = fmaxf(mx, logitsL[y]);
        exL[tid] = expf(logitsL[tid] - mx);
    }
    __syncthreads();
    if (tid < YDIM) {
        float s = 0.f;
        for (int y = 0; y < YDIM; ++y) s += exL[y];
        wvL[tid] = (exL[tid] / s) * validL[tid];
    }
    __syncthreads();
    if (tid < YDIM) {
        float d = 0.f;
        for (int y = 0; y < YDIM; ++y) d += wvL[y];
        d = fmaxf(d, 1e-6f);
        wtL[tid] = wvL[tid] / d;
    }
    __syncthreads();

    // ---- Phase 4: out[c] = sum_y tile_f32[y][c] * w[y]  (fp32 precision, from LDS) ----
    float acc = 0.f;
    #pragma unroll 8
    for (int y = 0; y < YDIM; ++y)
        acc = fmaf(tile[y * TSTRIDE + tid], wtL[y], acc);
    out[(size_t)bn * CDIM + tid] = acc;
}

extern "C" void kernel_launch(void* const* d_in, const int* in_sizes, int n_in,
                              void* d_out, int out_size, void* d_ws, size_t ws_size,
                              hipStream_t stream) {
    const float* sampled = (const float*)d_in[0];
    const float* valid   = (const float*)d_in[1];
    const float* ynorm   = (const float*)d_in[2];
    const float* W1      = (const float*)d_in[3];
    const float* b1      = (const float*)d_in[4];
    const float* W2      = (const float*)d_in[5];
    const float* b2      = (const float*)d_in[6];
    float* out = (float*)d_out;

    dim3 grid(BDIM * NDIM);   // 4096 token workgroups
    dim3 block(256);          // 8 wave32
    datok_kernel<<<grid, block, SMEM_BYTES, stream>>>(sampled, valid, ynorm,
                                                      W1, b1, W2, b2, out);
}
